// FactorizedReduce_39444979646686
// MI455X (gfx1250) — 8-bit matrix operand order repaired and checked by reading, compiled, not run on hardware
//
#include <hip/hip_runtime.h>
#include <hip/hip_bf16.h>
#include <stdint.h>

typedef int v8i __attribute__((ext_vector_type(8)));

#define BATCH   32
#define CIN     256
#define HIN     56
#define WIN     56
#define PLANE   (HIN * WIN)
#define HOUT    28
#define WOUT    28
#define POUT    (HOUT * WOUT)
#define MROWS   (BATCH * POUT)
#define NOUT    256
#define KDIM    256
#define COUT    512

__global__ void __launch_bounds__(256) bn_stats_kernel(const float* __restrict__ x,
                                                       float* __restrict__ mean,
                                                       float* __restrict__ invstd) {
    const int c = blockIdx.x;
    float s = 0.f, s2 = 0.f;
    for (int b = 0; b < BATCH; ++b) {
        const float4* p = (const float4*)(x + ((size_t)b * CIN + c) * PLANE);
        for (int i = threadIdx.x; i < PLANE / 4; i += 256) {
            const float4 v = p[i];
            s  += v.x + v.y + v.z + v.w;
            s2 += v.x * v.x + v.y * v.y + v.z * v.z + v.w * v.w;
        }
    }
    __shared__ float ss[256];
    __shared__ float ss2[256];
    ss[threadIdx.x]  = s;
    ss2[threadIdx.x] = s2;
    __syncthreads();
    for (int off = 128; off > 0; off >>= 1) {
        if (threadIdx.x < off) {
            ss[threadIdx.x]  += ss[threadIdx.x + off];
            ss2[threadIdx.x] += ss2[threadIdx.x + off];
        }
        __syncthreads();
    }
    if (threadIdx.x == 0) {
        const float inv_n = 1.0f / (float)(BATCH * PLANE);
        const float m   = ss[0] * inv_n;
        const float var = ss2[0] * inv_n - m * m;
        mean[c]   = m;
        invstd[c] = rsqrtf(var + 1e-5f);
    }
}

__global__ void __launch_bounds__(256) pack_w_kernel(const float* __restrict__ w1,
                                                     const float* __restrict__ w2,
                                                     uint32_t* __restrict__ b1t,
                                                     uint32_t* __restrict__ b2t) {
    const int i = blockIdx.x * 256 + threadIdx.x;
    const int base = i * 4;
    uint32_t p1 = 0, p2 = 0;
#pragma unroll
    for (int j = 0; j < 4; ++j) {
        const uint32_t s1 = (w1[base + j] >= 0.f) ? 0x01u : 0xFFu;
        const uint32_t s2 = (w2[base + j] >= 0.f) ? 0x01u : 0xFFu;
        p1 |= s1 << (8 * j);
        p2 |= s2 << (8 * j);
    }
    b1t[i] = p1;
    b2t[i] = p2;
}

__global__ void __launch_bounds__(256) pack_act_kernel(const float* __restrict__ x,
                                                       const float* __restrict__ mean,
                                                       const float* __restrict__ invstd,
                                                       const float* __restrict__ gamma,
                                                       const float* __restrict__ beta,
                                                       uint32_t* __restrict__ a1,
                                                       uint32_t* __restrict__ a2) {
    const int w  = blockIdx.x * 4 + (threadIdx.x >> 6);
    const int h  = blockIdx.y;
    const int b  = blockIdx.z;
    const int cg = threadIdx.x & 63;
    const int c0 = cg << 2;
    const int m  = (b * HOUT + h) * WOUT + w;

    uint32_t p1 = 0, p2 = 0;
#pragma unroll
    for (int j = 0; j < 4; ++j) {
        const int c = c0 + j;
        const size_t off1 = (((size_t)b * CIN + c) * HIN + 2 * h) * WIN + 2 * w;
        const size_t off2 = off1 + WIN + 1;
        if (c + 4 < CIN) __builtin_prefetch(&x[off1 + 4 * (size_t)PLANE], 0, 0);
        const float sc = invstd[c] * gamma[c];
        const float mn = mean[c];
        const float bt = beta[c];
        const float x1 = (x[off1] - mn) * sc + bt;
        const float x2 = (x[off2] - mn) * sc + bt;
        p1 |= ((x1 >= 0.f) ? 0x01u : 0xFFu) << (8 * j);
        p2 |= ((x2 >= 0.f) ? 0x01u : 0xFFu) << (8 * j);
    }
    a1[(size_t)m * 64 + cg] = p1;
    a2[(size_t)m * 64 + cg] = p2;
}

__global__ void __launch_bounds__(256) bgemm_wmma_kernel(const uint8_t* __restrict__ a1,
                                                         const uint8_t* __restrict__ a2,
                                                         const uint8_t* __restrict__ b1t,
                                                         const uint8_t* __restrict__ b2t,
                                                         float* __restrict__ out) {
    const int wave = threadIdx.x >> 5;
    const int lane = threadIdx.x & 31;
    const int sel  = blockIdx.y;
    const int unit = blockIdx.x * 8 + wave;
    const int ntg   = unit & 3;
    const int mtile = unit >> 2;

    const int b   = mtile / 49;
    const int hw0 = (mtile - b * 49) * 16;

    const uint8_t* __restrict__ A = sel ? a2 : a1;
    const uint8_t* __restrict__ B = sel ? b2t : b1t;

    const int half = lane >> 4;
    const int l15  = lane & 15;
    const uint8_t* arow  = A + (size_t)(mtile * 16 + l15) * KDIM;
    const uint8_t* brow0 = B + (size_t)(ntg * 64 + l15) * KDIM;

    v8i acc[4] = {};
#pragma unroll
    for (int kb = 0; kb < 4; ++kb) {
        const int kbase = kb * 64;
        v8i af;
#pragma unroll
        for (int v = 0; v < 8; ++v) {
            const int ka = kbase + ((v >> 1) << 4) + (half << 3) + ((v & 1) << 2);
            af[v] = *(const int*)(arow + ka);
        }
#pragma unroll
        for (int nt = 0; nt < 4; ++nt) {
            v8i bf;
#pragma unroll
            for (int v = 0; v < 8; ++v) {
                const int kn = kbase + ((v >> 1) << 4) + (half << 3) + ((v & 1) << 2);
                bf[v] = *(const int*)(brow0 + (size_t)nt * 16 * KDIM + kn);
            }
            acc[nt] = __builtin_amdgcn_wmma_i32_16x16x64_iu8(true, af, true, bf,
                                                             acc[nt], false, false);
        }
    }

#pragma unroll
    for (int nt = 0; nt < 4; ++nt) {
        const int oc = sel * NOUT + ntg * 64 + nt * 16 + l15;
        float* dst = out + ((size_t)b * COUT + oc) * POUT + hw0 + (half << 3);
        float4 lo, hi;
        lo.x = (float)acc[nt][0]; lo.y = (float)acc[nt][1];
        lo.z = (float)acc[nt][2]; lo.w = (float)acc[nt][3];
        hi.x = (float)acc[nt][4]; hi.y = (float)acc[nt][5];
        hi.z = (float)acc[nt][6]; hi.w = (float)acc[nt][7];
        *(float4*)(dst)     = lo;
        *(float4*)(dst + 4) = hi;
    }
}

extern "C" void kernel_launch(void* const* d_in, const int* in_sizes, int n_in,
                              void* d_out, int out_size, void* d_ws, size_t ws_size,
                              hipStream_t stream) {
    const float* x     = (const float*)d_in[0];
    const float* gamma = (const float*)d_in[1];
    const float* beta  = (const float*)d_in[2];
    const float* w1    = (const float*)d_in[3];
    const float* w2    = (const float*)d_in[4];
    float* out = (float*)d_out;

    uint8_t* ws = (uint8_t*)d_ws;
    float*    mean   = (float*)(ws);
    float*    invstd = (float*)(ws + 1024);
    uint8_t*  b1t    = ws + 2048;
    uint8_t*  b2t    = b1t + (size_t)NOUT * KDIM;
    uint8_t*  a1     = b2t + (size_t)NOUT * KDIM;
    uint8_t*  a2     = a1 + (size_t)MROWS * KDIM;

    bn_stats_kernel<<<CIN, 256, 0, stream>>>(x, mean, invstd);

    pack_w_kernel<<<(NOUT * KDIM / 4) / 256, 256, 0, stream>>>(
        w1, w2, (uint32_t*)b1t, (uint32_t*)b2t);

    pack_act_kernel<<<dim3(WOUT / 4, HOUT, BATCH), 256, 0, stream>>>(
        x, mean, invstd, gamma, beta, (uint32_t*)a1, (uint32_t*)a2);

    bgemm_wmma_kernel<<<dim3(784, 2), 256, 0, stream>>>(a1, a2, b1t, b2t, out);
}
